// TransformerMPNN_23381801960100
// MI455X (gfx1250) — matrix operand order repaired and checked by reading, not compiled, not run on hardware
//
#include <hip/hip_runtime.h>
#include <hip/hip_bf16.h>

// ---------------------------------------------------------------------------
// Types for CDNA5 WMMA (wave32): A/B = 16 bf16 (8 VGPRs), C/D = 8 f32
// ---------------------------------------------------------------------------
typedef __attribute__((ext_vector_type(16))) __bf16 v16bf;
typedef __attribute__((ext_vector_type(8)))  __bf16 v8bf;
typedef __attribute__((ext_vector_type(8)))  float  v8f;

#define WMMA_BF16(a, b, c) \
    __builtin_amdgcn_wmma_f32_16x16x32_bf16(false, (a), false, (b), (short)0, (c), false, false)

__device__ __forceinline__ unsigned short f2bf(float x) {
    unsigned int u = __float_as_uint(x);
    unsigned int r = u + 0x7FFFu + ((u >> 16) & 1u);   // round-to-nearest-even
    return (unsigned short)(r >> 16);
}

// A fragment: 16x32 bf16 tile, row-major source, lane=(hi<<4)|m.
//   elems 0..7  = A[m][kk + 8*hi + 0..7]
//   elems 8..15 = A[m][kk + 16 + 8*hi + 0..7]
__device__ __forceinline__ v16bf load_afrag(const unsigned short* __restrict__ A,
                                            int lda, int kk, int m, int hi) {
    union { v16bf v; v8bf h[2]; } u;
    const unsigned short* p = A + (size_t)m * lda + kk + 8 * hi;
    u.h[0] = *(const v8bf*)(const void*)(p);
    u.h[1] = *(const v8bf*)(const void*)(p + 16);
    return u.v;
}

// B fragment: 32x16 bf16 tile from Bt (row n of Bt == column n of B), lane=(hi<<4)|n.
//   elems 0..7  = Bt[n][kk + 8*hi + 0..7]
//   elems 8..15 = Bt[n][kk + 16 + 8*hi + 0..7]
__device__ __forceinline__ v16bf load_bfrag(const unsigned short* __restrict__ Bt,
                                            int ldb, int kk, int n, int hi) {
    union { v16bf v; v8bf h[2]; } u;
    const unsigned short* p = Bt + (size_t)n * ldb + kk + 8 * hi;
    u.h[0] = *(const v8bf*)(const void*)(p);
    u.h[1] = *(const v8bf*)(const void*)(p + 16);
    return u.v;
}

// ---------------------------------------------------------------------------
// Small prep kernels
// ---------------------------------------------------------------------------
__global__ void k_f2bf(unsigned short* __restrict__ dst, const float* __restrict__ src, int n) {
    int i = blockIdx.x * blockDim.x + threadIdx.x;
    if (i < n) dst[i] = f2bf(src[i]);
}

// src: [R, C] f32 row-major  ->  dst: [C, R] bf16 row-major (i.e. B^T for GEMM)
__global__ void k_transbf(unsigned short* __restrict__ dst, const float* __restrict__ src,
                          int R, int C) {
    int i = blockIdx.x * blockDim.x + threadIdx.x;
    if (i >= R * C) return;
    int r = i / C, c = i % C;
    dst[(size_t)c * R + r] = f2bf(src[i]);
}

__global__ void k_deg_init(float* __restrict__ deg, int N) {
    int i = blockIdx.x * blockDim.x + threadIdx.x;
    if (i < N) deg[i] = 1.0f;                    // self-loop weight
}

__global__ void k_deg_scatter(float* __restrict__ deg, const int* __restrict__ ei,
                              const float* __restrict__ ew, int E) {
    int e = blockIdx.x * blockDim.x + threadIdx.x;
    if (e < E) atomicAdd(&deg[ei[E + e]], ew[e]);  // dst = ei[1][e]
}

__global__ void k_dinv(float* __restrict__ dinv, const float* __restrict__ deg, int N) {
    int i = blockIdx.x * blockDim.x + threadIdx.x;
    if (i >= N) return;
    float d = deg[i];
    dinv[i] = (d > 0.f) ? rsqrtf(fmaxf(d, 1e-12f)) : 0.f;
}

// gcn[n][d] = bg[d] + dinv[n]^2 * xt[n][d]   (self-loop contribution + bias)
__global__ void k_gcn_self(float* __restrict__ gcn, const float* __restrict__ xt,
                           const float* __restrict__ dinv, const float* __restrict__ bg, int N) {
    int i = blockIdx.x * blockDim.x + threadIdx.x;
    if (i >= N * 512) return;
    int n = i >> 9, d = i & 511;
    float di = dinv[n];
    gcn[i] = bg[d] + di * di * xt[i];
}

// one block per edge, 128 threads x 4 floats = 512 columns
__global__ void __launch_bounds__(128)
k_gcn_edges(float* __restrict__ gcn, const float* __restrict__ xt,
            const float* __restrict__ dinv, const int* __restrict__ ei,
            const float* __restrict__ ew, int E) {
    int e = blockIdx.x;
    int src = ei[e], dst = ei[E + e];
    float coef = dinv[src] * ew[e] * dinv[dst];
    const float* xs = xt + (size_t)src * 512;
    float* gd = gcn + (size_t)dst * 512;
    int d = threadIdx.x * 4;
    atomicAdd(&gd[d + 0], coef * xs[d + 0]);
    atomicAdd(&gd[d + 1], coef * xs[d + 1]);
    atomicAdd(&gd[d + 2], coef * xs[d + 2]);
    atomicAdd(&gd[d + 3], coef * xs[d + 3]);
}

// vT[h*64+dh][n] = v[n][h*64+dh]  (per-head V^T for the P@V WMMA B operand)
__global__ void k_vtrans(unsigned short* __restrict__ vT, const unsigned short* __restrict__ v,
                         int N) {
    int i = blockIdx.x * blockDim.x + threadIdx.x;
    if (i >= N * 512) return;
    int n = i >> 9, c = i & 511;
    vT[(size_t)c * N + n] = v[i];
}

// ---------------------------------------------------------------------------
// LayerNorm over D=512, one wave per row, output bf16
// ---------------------------------------------------------------------------
__global__ void __launch_bounds__(128)
k_ln(unsigned short* __restrict__ out, const float* __restrict__ x,
     const float* __restrict__ s, const float* __restrict__ b, int N) {
    int lane = threadIdx.x & 31, w = threadIdx.x >> 5;
    int row = blockIdx.x * 4 + w;
    if (row >= N) return;
    const float* xr = x + (size_t)row * 512;
    float4 v[4];
    float sum = 0.f, sq = 0.f;
#pragma unroll
    for (int i = 0; i < 4; ++i) {
        v[i] = *(const float4*)(const void*)(xr + lane * 16 + i * 4);
        sum += v[i].x + v[i].y + v[i].z + v[i].w;
        sq  += v[i].x * v[i].x + v[i].y * v[i].y + v[i].z * v[i].z + v[i].w * v[i].w;
    }
#pragma unroll
    for (int m = 16; m >= 1; m >>= 1) {
        sum += __shfl_xor(sum, m, 32);
        sq  += __shfl_xor(sq, m, 32);
    }
    float mean = sum * (1.0f / 512.0f);
    float var  = sq * (1.0f / 512.0f) - mean * mean;
    float inv  = rsqrtf(var + 1e-5f);
#pragma unroll
    for (int i = 0; i < 4; ++i) {
        int c0 = lane * 16 + i * 4;
        float o0 = (v[i].x - mean) * inv * s[c0 + 0] + b[c0 + 0];
        float o1 = (v[i].y - mean) * inv * s[c0 + 1] + b[c0 + 1];
        float o2 = (v[i].z - mean) * inv * s[c0 + 2] + b[c0 + 2];
        float o3 = (v[i].w - mean) * inv * s[c0 + 3] + b[c0 + 3];
        out[(size_t)row * 512 + c0 + 0] = f2bf(o0);
        out[(size_t)row * 512 + c0 + 1] = f2bf(o1);
        out[(size_t)row * 512 + c0 + 2] = f2bf(o2);
        out[(size_t)row * 512 + c0 + 3] = f2bf(o3);
    }
}

// ---------------------------------------------------------------------------
// Register-blocked WMMA GEMM:  C[M,Nc] = act(A[M,K] @ Bt[Nc,K]^T + bias) + resid
//   Each wave computes a 32x64 output tile: 2 A-frags x 4 B-frags -> 8 WMMAs
//   per 32-wide K step (6KB fragment traffic per 128K FLOPs).
//   Block = 4 waves stacked along M: block tile = 128M x 64N.
// ---------------------------------------------------------------------------
__global__ void __launch_bounds__(128)
k_gemm(const unsigned short* __restrict__ A, const unsigned short* __restrict__ Bt,
       const float* __restrict__ bias, const float* __restrict__ resid,
       float* __restrict__ Cf, unsigned short* __restrict__ Cbf,
       int M, int Ncols, int K, int actGelu) {
    int lane = threadIdx.x & 31, w = threadIdx.x >> 5;
    int tn = blockIdx.x * 64;                 // block N base (64 cols)
    int tm = blockIdx.y * 128 + w * 32;       // wave M base (32 rows)
    if (tn >= Ncols || tm >= M) return;       // uniform per wave
    int m = lane & 15, hi = lane >> 4;
    const unsigned short* A0 = A + (size_t)tm * K;
    const unsigned short* A1 = A + (size_t)(tm + 16) * K;
    const unsigned short* B0 = Bt + (size_t)tn * K;

    v8f z = {};
    v8f acc[2][4];
#pragma unroll
    for (int i = 0; i < 2; ++i)
#pragma unroll
        for (int j = 0; j < 4; ++j) acc[i][j] = z;

    for (int kk = 0; kk < K; kk += 32) {
        if (kk + 64 < K) {
            __builtin_prefetch(A0 + (size_t)m * K + kk + 64, 0, 3);
            __builtin_prefetch(A1 + (size_t)m * K + kk + 64, 0, 3);
        }
        v16bf a0 = load_afrag(A0, K, kk, m, hi);
        v16bf a1 = load_afrag(A1, K, kk, m, hi);
        v16bf b0 = load_bfrag(B0, K, kk, m, hi);
        v16bf b1 = load_bfrag(B0 + (size_t)16 * K, K, kk, m, hi);
        v16bf b2 = load_bfrag(B0 + (size_t)32 * K, K, kk, m, hi);
        v16bf b3 = load_bfrag(B0 + (size_t)48 * K, K, kk, m, hi);
        acc[0][0] = WMMA_BF16(a0, b0, acc[0][0]);
        acc[0][1] = WMMA_BF16(a0, b1, acc[0][1]);
        acc[0][2] = WMMA_BF16(a0, b2, acc[0][2]);
        acc[0][3] = WMMA_BF16(a0, b3, acc[0][3]);
        acc[1][0] = WMMA_BF16(a1, b0, acc[1][0]);
        acc[1][1] = WMMA_BF16(a1, b1, acc[1][1]);
        acc[1][2] = WMMA_BF16(a1, b2, acc[1][2]);
        acc[1][3] = WMMA_BF16(a1, b3, acc[1][3]);
    }

#pragma unroll
    for (int i = 0; i < 2; ++i) {
#pragma unroll
        for (int j = 0; j < 4; ++j) {
            int col = tn + j * 16 + m;
            float bc = bias ? bias[col] : 0.0f;
#pragma unroll
            for (int r = 0; r < 8; ++r) {
                int row = tm + i * 16 + r + 8 * hi;
                float v = acc[i][j][r] + bc;
                if (actGelu) {
                    float t = v;
                    v = 0.5f * t * (1.0f + tanhf(0.7978845608028654f * (t + 0.044715f * t * t * t)));
                }
                size_t o = (size_t)row * Ncols + col;
                if (resid) v += resid[o];
                if (Cf)  Cf[o] = v;
                if (Cbf) Cbf[o] = f2bf(v);
            }
        }
    }
}

// ---------------------------------------------------------------------------
// Flash attention: one wave per (16-query tile, head). N=4096, DH=64, H=8.
//   scores: 4 WMMAs / 32-key block; online softmax; P restaged via LDS;
//   O += P @ V^T: 4 WMMAs / block.
// ---------------------------------------------------------------------------
__global__ void __launch_bounds__(128)
k_attn(const unsigned short* __restrict__ q, const unsigned short* __restrict__ kmat,
       const unsigned short* __restrict__ vT, unsigned short* __restrict__ obuf, int N) {
    __shared__ alignas(16) unsigned short Plds[4][16 * 32];
    int lane = threadIdx.x & 31, w = threadIdx.x >> 5;
    int qt = blockIdx.x * 4 + w;
    int h = blockIdx.y;
    int q0 = qt * 16;
    int m = lane & 15, hi = lane >> 4;
    const int D = 512;
    unsigned short* myP = Plds[w];

    const unsigned short* qrow = q + (size_t)q0 * D + h * 64;
    v16bf aq0 = load_afrag(qrow, D, 0, m, hi);
    v16bf aq1 = load_afrag(qrow, D, 32, m, hi);

    float mst[8], lst[8];
    v8f o0 = {}, o1 = {}, o2 = {}, o3 = {};
#pragma unroll
    for (int r = 0; r < 8; ++r) { mst[r] = -1e30f; lst[r] = 0.f; }

    const unsigned short* vtb = vT + (size_t)h * 64 * N;

    for (int kb = 0; kb < N; kb += 32) {
        const unsigned short* kr0 = kmat + (size_t)kb * D + h * 64;
        const unsigned short* kr1 = kmat + (size_t)(kb + 16) * D + h * 64;
        v16bf b00 = load_bfrag(kr0, D, 0, m, hi);
        v16bf b01 = load_bfrag(kr0, D, 32, m, hi);
        v16bf b10 = load_bfrag(kr1, D, 0, m, hi);
        v16bf b11 = load_bfrag(kr1, D, 32, m, hi);
        v8f z = {};
        v8f s0 = WMMA_BF16(aq0, b00, z);
        s0 = WMMA_BF16(aq1, b01, s0);
        v8f s1 = WMMA_BF16(aq0, b10, z);
        s1 = WMMA_BF16(aq1, b11, s1);

        float corr[8];
#pragma unroll
        for (int r = 0; r < 8; ++r) {
            float a = s0[r] * 0.125f, bb = s1[r] * 0.125f;  // 1/sqrt(64)
            float rm = fmaxf(a, bb);
            rm = fmaxf(rm, __shfl_xor(rm, 1, 32));
            rm = fmaxf(rm, __shfl_xor(rm, 2, 32));
            rm = fmaxf(rm, __shfl_xor(rm, 4, 32));
            rm = fmaxf(rm, __shfl_xor(rm, 8, 32));
            float mn = fmaxf(mst[r], rm);
            corr[r] = __expf(mst[r] - mn);
            float p0 = __expf(a - mn), p1 = __expf(bb - mn);
            float rs = p0 + p1;
            rs += __shfl_xor(rs, 1, 32);
            rs += __shfl_xor(rs, 2, 32);
            rs += __shfl_xor(rs, 4, 32);
            rs += __shfl_xor(rs, 8, 32);
            lst[r] = lst[r] * corr[r] + rs;
            mst[r] = mn;
            int row = r + 8 * hi;                 // C layout: elem r -> row r+8*hi, col n
            myP[row * 32 + m]      = f2bf(p0);
            myP[row * 32 + 16 + m] = f2bf(p1);
        }
        // wave-internal LDS RAW: explicit CDNA5 split-counter wait
        asm volatile("s_wait_dscnt 0" ::: "memory");

        // reload P as a 16x32 A fragment
        union { v16bf v; v8bf hv[2]; } pu;
        pu.hv[0] = *(const v8bf*)(const void*)(myP + m * 32 + 8 * hi);
        pu.hv[1] = *(const v8bf*)(const void*)(myP + m * 32 + 16 + 8 * hi);
        v16bf pf = pu.v;

#pragma unroll
        for (int r = 0; r < 8; ++r) {
            o0[r] *= corr[r]; o1[r] *= corr[r]; o2[r] *= corr[r]; o3[r] *= corr[r];
        }
        v16bf bv0 = load_bfrag(vtb + (size_t)0  * N, N, kb, m, hi);
        v16bf bv1 = load_bfrag(vtb + (size_t)16 * N, N, kb, m, hi);
        v16bf bv2 = load_bfrag(vtb + (size_t)32 * N, N, kb, m, hi);
        v16bf bv3 = load_bfrag(vtb + (size_t)48 * N, N, kb, m, hi);
        o0 = WMMA_BF16(pf, bv0, o0);
        o1 = WMMA_BF16(pf, bv1, o1);
        o2 = WMMA_BF16(pf, bv2, o2);
        o3 = WMMA_BF16(pf, bv3, o3);
    }

#pragma unroll
    for (int r = 0; r < 8; ++r) {
        float inv = (lst[r] > 0.f) ? 1.0f / lst[r] : 0.f;
        int row = q0 + r + 8 * hi;
        size_t base = (size_t)row * 512 + h * 64;
        obuf[base + 0  + m] = f2bf(o0[r] * inv);
        obuf[base + 16 + m] = f2bf(o1[r] * inv);
        obuf[base + 32 + m] = f2bf(o2[r] * inv);
        obuf[base + 48 + m] = f2bf(o3[r] * inv);
    }
}

// ---------------------------------------------------------------------------
// Host launcher
// ---------------------------------------------------------------------------
extern "C" void kernel_launch(void* const* d_in, const int* in_sizes, int n_in,
                              void* d_out, int out_size, void* d_ws, size_t ws_size,
                              hipStream_t stream) {
    (void)in_sizes; (void)n_in; (void)out_size; (void)ws_size;
    const int N = 4096, D = 512, H = 8, DFF = 2048, E = 65536;

    const float* x    = (const float*)d_in[0];
    const int*   ei   = (const int*)d_in[1];
    const float* ew   = (const float*)d_in[2];
    const float* Wg   = (const float*)d_in[3];
    const float* bg   = (const float*)d_in[4];
    const float* ln1s = (const float*)d_in[5];
    const float* ln1b = (const float*)d_in[6];
    const float* Wq   = (const float*)d_in[7];
    const float* bq   = (const float*)d_in[8];
    const float* Wk   = (const float*)d_in[9];
    const float* bk   = (const float*)d_in[10];
    const float* Wv   = (const float*)d_in[11];
    const float* bv   = (const float*)d_in[12];
    const float* Wo   = (const float*)d_in[13];
    const float* bo   = (const float*)d_in[14];
    const float* ln2s = (const float*)d_in[15];
    const float* ln2b = (const float*)d_in[16];
    const float* W1   = (const float*)d_in[17];
    const float* b1   = (const float*)d_in[18];
    const float* W2   = (const float*)d_in[19];
    const float* b2   = (const float*)d_in[20];
    const float* Wa   = (const float*)d_in[21];
    const float* ba   = (const float*)d_in[22];
    float* out = (float*)d_out;

    // bump allocator over workspace
    char* ws = (char*)d_ws;
    size_t off = 0;
    auto alloc = [&](size_t bytes) -> void* {
        void* p = ws + off;
        off = (off + bytes + 255) & ~(size_t)255;
        return p;
    };
    typedef unsigned short u16;
    u16* WgT   = (u16*)alloc((size_t)D * D * 2);
    u16* WqT   = (u16*)alloc((size_t)D * D * 2);
    u16* WkT   = (u16*)alloc((size_t)D * D * 2);
    u16* WvT   = (u16*)alloc((size_t)D * D * 2);
    u16* WoT   = (u16*)alloc((size_t)D * D * 2);
    u16* W1T   = (u16*)alloc((size_t)D * DFF * 2);
    u16* W2T   = (u16*)alloc((size_t)DFF * D * 2);
    u16* WaTT  = (u16*)alloc((size_t)D * D * 2);
    u16* WaBT  = (u16*)alloc((size_t)D * D * 2);
    u16* x_bf  = (u16*)alloc((size_t)N * D * 2);
    u16* h1_bf = (u16*)alloc((size_t)N * D * 2);
    u16* q_bf  = (u16*)alloc((size_t)N * D * 2);
    u16* k_bf  = (u16*)alloc((size_t)N * D * 2);
    u16* v_bf  = (u16*)alloc((size_t)N * D * 2);
    u16* vT_bf = (u16*)alloc((size_t)N * D * 2);
    u16* o_bf  = (u16*)alloc((size_t)N * D * 2);
    u16* h2_bf = (u16*)alloc((size_t)N * D * 2);
    u16* mid_bf= (u16*)alloc((size_t)N * DFF * 2);
    u16* tfm_bf= (u16*)alloc((size_t)N * D * 2);
    u16* gcn_bf= (u16*)alloc((size_t)N * D * 2);
    float* xt_f  = (float*)alloc((size_t)N * D * 4);
    float* gcn_f = (float*)alloc((size_t)N * D * 4);
    float* x2_f  = (float*)alloc((size_t)N * D * 4);
    float* tfm_f = (float*)alloc((size_t)N * D * 4);
    float* tmp_f = (float*)alloc((size_t)N * D * 4);
    float* deg   = (float*)alloc((size_t)N * 4);
    float* dinv  = (float*)alloc((size_t)N * 4);

    auto cdiv = [](int a, int b) { return (a + b - 1) / b; };

    // --- weight conversion (f32 -> bf16, transposed) ---
    k_transbf<<<cdiv(D * D, 256), 256, 0, stream>>>(WgT, Wg, D, D);
    k_transbf<<<cdiv(D * D, 256), 256, 0, stream>>>(WqT, Wq, D, D);
    k_transbf<<<cdiv(D * D, 256), 256, 0, stream>>>(WkT, Wk, D, D);
    k_transbf<<<cdiv(D * D, 256), 256, 0, stream>>>(WvT, Wv, D, D);
    k_transbf<<<cdiv(D * D, 256), 256, 0, stream>>>(WoT, Wo, D, D);
    k_transbf<<<cdiv(D * DFF, 256), 256, 0, stream>>>(W1T, W1, D, DFF);
    k_transbf<<<cdiv(DFF * D, 256), 256, 0, stream>>>(W2T, W2, DFF, D);
    k_transbf<<<cdiv(D * D, 256), 256, 0, stream>>>(WaTT, Wa, D, D);
    k_transbf<<<cdiv(D * D, 256), 256, 0, stream>>>(WaBT, Wa + (size_t)D * D, D, D);
    k_f2bf<<<cdiv(N * D, 256), 256, 0, stream>>>(x_bf, x, N * D);

    dim3 blk(128);
    dim3 g512(D / 64, N / 128);     // Ncols=512 GEMMs: 8 x 32 blocks
    dim3 gdff(DFF / 64, N / 128);   // Ncols=2048 GEMM: 32 x 32 blocks

    // --- GCN path ---
    k_deg_init<<<cdiv(N, 256), 256, 0, stream>>>(deg, N);
    k_deg_scatter<<<cdiv(E, 256), 256, 0, stream>>>(deg, ei, ew, E);
    k_dinv<<<cdiv(N, 256), 256, 0, stream>>>(dinv, deg, N);
    k_gemm<<<g512, blk, 0, stream>>>(x_bf, WgT, nullptr, nullptr, xt_f, nullptr, N, D, D, 0);
    k_gcn_self<<<cdiv(N * D, 256), 256, 0, stream>>>(gcn_f, xt_f, dinv, bg, N);
    k_gcn_edges<<<E, blk, 0, stream>>>(gcn_f, xt_f, dinv, ei, ew, E);
    k_f2bf<<<cdiv(N * D, 256), 256, 0, stream>>>(gcn_bf, gcn_f, N * D);

    // --- transformer path ---
    k_ln<<<N / 4, blk, 0, stream>>>(h1_bf, x, ln1s, ln1b, N);
    k_gemm<<<g512, blk, 0, stream>>>(h1_bf, WqT, bq, nullptr, nullptr, q_bf, N, D, D, 0);
    k_gemm<<<g512, blk, 0, stream>>>(h1_bf, WkT, bk, nullptr, nullptr, k_bf, N, D, D, 0);
    k_gemm<<<g512, blk, 0, stream>>>(h1_bf, WvT, bv, nullptr, nullptr, v_bf, N, D, D, 0);
    k_vtrans<<<cdiv(N * D, 256), 256, 0, stream>>>(vT_bf, v_bf, N);
    k_attn<<<dim3(N / 64, H), blk, 0, stream>>>(q_bf, k_bf, vT_bf, o_bf, N);
    k_gemm<<<g512, blk, 0, stream>>>(o_bf, WoT, bo, x, x2_f, nullptr, N, D, D, 0);
    k_ln<<<N / 4, blk, 0, stream>>>(h2_bf, x2_f, ln2s, ln2b, N);
    k_gemm<<<gdff, blk, 0, stream>>>(h2_bf, W1T, b1, nullptr, nullptr, mid_bf, N, DFF, D, 1);
    k_gemm<<<g512, blk, 0, stream>>>(mid_bf, W2T, b2, x2_f, tfm_f, tfm_bf, N, D, DFF, 0);

    // --- output projection: out = x + gcn@WaTop + tfm@WaBot + ba ---
    k_gemm<<<g512, blk, 0, stream>>>(gcn_bf, WaTT, ba, x, tmp_f, nullptr, N, D, D, 0);
    k_gemm<<<g512, blk, 0, stream>>>(tfm_bf, WaBT, nullptr, tmp_f, out, nullptr, N, D, D, 0);
    (void)H;
}
